// CDConvBlock_28913719836782
// MI455X (gfx1250) — matrix operand order repaired and checked by reading, not compiled, not run on hardware
//
#include <hip/hip_runtime.h>
#include <hip/hip_bf16.h>

#define Bn 8
#define Cn 256
#define Hn 128
#define Wn 128
#define HWn (Hn * Wn)

typedef __attribute__((ext_vector_type(16))) __bf16 v16bf;
typedef __attribute__((ext_vector_type(8)))  float  v8f;
typedef __attribute__((ext_vector_type(4)))  unsigned int u32x4;
typedef __attribute__((ext_vector_type(2)))  unsigned int u32x2;

union FragBF { v16bf v; u32x4 q[2]; };

__device__ __forceinline__ unsigned short f2bf(float f) {
  unsigned u = __builtin_bit_cast(unsigned, f);
  u += 0x7FFFu + ((u >> 16) & 1u);   // round-to-nearest-even
  return (unsigned short)(u >> 16);
}

// ---------------------------------------------------------------------------
// Kernel 1: W2[o][c] = (sum_k w_W[o][k] * w_pc[k][c]) * w_dw[c]  -> bf16
// ---------------------------------------------------------------------------
__global__ void cdcb_precompute_w2(const float* __restrict__ wW,
                                   const float* __restrict__ wPc,
                                   const float* __restrict__ wDw,
                                   unsigned short* __restrict__ W2bf) {
  const int o = blockIdx.x;
  const int c = threadIdx.x;
  float acc = 0.f;
#pragma unroll 4
  for (int k = 0; k < Cn; ++k)
    acc += wW[o * Cn + k] * wPc[k * Cn + c];
  W2bf[o * Cn + c] = f2bf(acc * wDw[c]);
}

// ---------------------------------------------------------------------------
// Kernel 2: per-pixel sampling parameters (ix0, iy0, wx, wy).
// 4 pixels per thread, float4 loads over the channel reduction.
// ---------------------------------------------------------------------------
__global__ void cdcb_offsets(const float* __restrict__ x,
                             const float* __restrict__ wOff,
                             const float* __restrict__ bOff,
                             float4* __restrict__ prm) {
  const int q4  = (blockIdx.x * 256 + threadIdx.x) * 4;  // 4 consecutive pixels
  const int b   = q4 >> 14;
  const int rem = q4 & (HWn - 1);
  const int h   = rem >> 7;
  const int w   = rem & (Wn - 1);                         // multiple of 4
  const float* xb = x + (size_t)b * Cn * HWn + rem;
  float a0[4] = {0.f, 0.f, 0.f, 0.f};
  float a1[4] = {0.f, 0.f, 0.f, 0.f};
#pragma unroll 4
  for (int c = 0; c < Cn; ++c) {
    float4 v = *(const float4*)(xb + (size_t)c * HWn);
    float vv[4] = {v.x, v.y, v.z, v.w};
    float w0 = wOff[c], w1 = wOff[Cn + c];
#pragma unroll
    for (int j = 0; j < 4; ++j) { a0[j] += w0 * vv[j]; a1[j] += w1 * vv[j]; }
  }
  const float b0 = bOff[0], b1 = bOff[1];
  const float gyb = -1.f + (float)h * (2.f / 127.f);
#pragma unroll
  for (int j = 0; j < 4; ++j) {
    float ox = tanhf(a0[j] + b0);
    float oy = tanhf(a1[j] + b1);
    float gx = fminf(fmaxf(-1.f + (float)(w + j) * (2.f / 127.f) + ox, -1.f), 1.f);
    float gy = fminf(fmaxf(gyb + oy, -1.f), 1.f);
    float ix = (gx + 1.f) * 64.f - 0.5f;   // ((gx+1)*W - 1)*0.5
    float iy = (gy + 1.f) * 64.f - 0.5f;
    float fx = floorf(ix), fy = floorf(iy);
    prm[q4 + j] = make_float4(fx, fy, ix - fx, iy - fy);
  }
}

// ---------------------------------------------------------------------------
// Kernel 3: gather + bilinear blend -> bf16 LDS tile -> 256x256 WMMA GEMM
//           out = W2 @ bilinear(x) + b_W + x
// Block: 64 pixels (one batch image), 512 threads = 16 waves.
// Each wave owns a 32(M) x 32(N) output tile: 2 A-frags x 2 B-frags -> 4 WMMA.
// ---------------------------------------------------------------------------
#define BSTRIDE 264   // 256 + 8 halfs padding

__global__ void __launch_bounds__(512)
cdcb_main(const float* __restrict__ x,
          const unsigned short* __restrict__ W2bf,
          const float4* __restrict__ prm,
          const float* __restrict__ bW,
          float* __restrict__ out) {
  __shared__ __align__(16) unsigned short smB[64 * BSTRIDE];
  __shared__ int   sA[2][64];   // pair-load base offsets (row y0, row y1)
  __shared__ float sW[4][64];   // weights for p0.x, p0.y, p1.x, p1.y

  const int tid   = threadIdx.x;
  const int pbase = blockIdx.x * 64;
  const int bImg  = pbase >> 14;
  const int pin   = pbase & (HWn - 1);

  // ---- Phase 1: per-pixel pair-load addresses + edge-folded weights ----
  if (tid < 64) {
    float4 pr = prm[pbase + tid];
    int ix0 = (int)pr.x, iy0 = (int)pr.y;           // ix0,iy0 in [-1, 127]
    float wx = pr.z, wy = pr.w;
    // x taps collapse to one float2 load at cxp = clamp(ix0, 0, W-2);
    // fold boundary selection into the pair weights (ax -> .x, bx -> .y).
    int cxp = min(max(ix0, 0), Wn - 2);
    float ax, bx;
    if (ix0 < 0)            { ax = wx;       bx = 0.f;      }  // ix0 == -1
    else if (ix0 > Wn - 2)  { ax = 0.f;      bx = 1.f - wx; }  // ix0 == 127
    else                    { ax = 1.f - wx; bx = wx;       }
    int iy1 = iy0 + 1;
    float ay = (iy0 >= 0) ? (1.f - wy) : 0.f;
    float by = (iy1 < Hn) ? wy : 0.f;
    int ry0 = max(iy0, 0);
    int ry1 = min(iy1, Hn - 1);
    sA[0][tid] = ry0 * Wn + cxp;
    sA[1][tid] = ry1 * Wn + cxp;
    sW[0][tid] = ay * ax;
    sW[1][tid] = ay * bx;
    sW[2][tid] = by * ax;
    sW[3][tid] = by * bx;
  }
  __syncthreads();

  // ---- Phase 2: gather (2 x b64 per channel) -> bf16 B-tile [N=64][K=256] ----
  {
    const int p = tid & 63;
    const int g = tid >> 6;                 // channel group 0..7, 32 ch each
    const int a0 = sA[0][p], a1 = sA[1][p];
    const float q0 = sW[0][p], q1 = sW[1][p], q2 = sW[2][p], q3 = sW[3][p];
    const float* xb = x + (size_t)bImg * Cn * HWn + (size_t)g * 32 * HWn;
    unsigned short* dst = &smB[p * BSTRIDE + g * 32];
#pragma unroll
    for (int j4 = 0; j4 < 32; j4 += 4) {
      float s[4];
#pragma unroll
      for (int t = 0; t < 4; ++t) {
        const float* bc = xb + (size_t)(j4 + t) * HWn;
        float2 p0, p1;
        __builtin_memcpy(&p0, bc + a0, 8);   // unaligned-ok 8B load
        __builtin_memcpy(&p1, bc + a1, 8);
        s[t] = q0 * p0.x + q1 * p0.y + q2 * p1.x + q3 * p1.y;
      }
      u32x2 pk;
      pk.x = (unsigned)f2bf(s[0]) | ((unsigned)f2bf(s[1]) << 16);
      pk.y = (unsigned)f2bf(s[2]) | ((unsigned)f2bf(s[3]) << 16);
      *(u32x2*)(dst + j4) = pk;              // ds_store_b64
    }
  }
  __syncthreads();

  // ---- Phase 3: GEMM. Wave (wy,wx) owns M [wy*32,+32) x N [wx*32,+32) ----
  const int wv = tid >> 5, lane = tid & 31;
  const int wgy = wv & 7;          // 8 M-groups
  const int wgx = wv >> 3;         // 2 N-groups
  const int lr = lane & 15, hi = lane >> 4;
  const unsigned short* Arow0 = W2bf + (wgy * 32 + lr) * Cn;
  const unsigned short* Arow1 = Arow0 + 16 * Cn;

  v8f acc00 = {0, 0, 0, 0, 0, 0, 0, 0};
  v8f acc01 = acc00, acc10 = acc00, acc11 = acc00;

#pragma unroll
  for (int k0 = 0; k0 < Cn; k0 += 32) {
    FragBF a0, a1;
    // A 16x32 bf16 layout: lane-lo K{0-7,16-23}, lane-hi K{8-15,24-31}
    a0.q[0] = *(const u32x4*)(Arow0 + k0 + hi * 8);
    a0.q[1] = *(const u32x4*)(Arow0 + k0 + 16 + hi * 8);
    a1.q[0] = *(const u32x4*)(Arow1 + k0 + hi * 8);
    a1.q[1] = *(const u32x4*)(Arow1 + k0 + 16 + hi * 8);
    // B 32x16 bf16 layout: lane-lo K{0-7,16-23}, lane-hi K{8-15,24-31}
    const unsigned short* bb = &smB[(wgx * 32 + lr) * BSTRIDE + k0 + hi * 8];
    FragBF b0, b1;
    b0.q[0] = *(const u32x4*)(bb);                 b0.q[1] = *(const u32x4*)(bb + 16);
    b1.q[0] = *(const u32x4*)(bb + 16 * BSTRIDE);  b1.q[1] = *(const u32x4*)(bb + 16 * BSTRIDE + 16);
    acc00 = __builtin_amdgcn_wmma_f32_16x16x32_bf16(false, a0.v, false, b0.v, (short)0, acc00, false, false);
    acc01 = __builtin_amdgcn_wmma_f32_16x16x32_bf16(false, a0.v, false, b1.v, (short)0, acc01, false, false);
    acc10 = __builtin_amdgcn_wmma_f32_16x16x32_bf16(false, a1.v, false, b0.v, (short)0, acc10, false, false);
    acc11 = __builtin_amdgcn_wmma_f32_16x16x32_bf16(false, a1.v, false, b1.v, (short)0, acc11, false, false);
  }

  // ---- Epilogue: + b_W + residual x.  D layout: lane-lo M=r, lane-hi M=8+r, N=lr
  float bw0[8], bw1[8];
#pragma unroll
  for (int r = 0; r < 8; ++r) {
    bw0[r] = bW[wgy * 32 + hi * 8 + r];
    bw1[r] = bW[wgy * 32 + 16 + hi * 8 + r];
  }
#pragma unroll
  for (int mt = 0; mt < 2; ++mt) {
#pragma unroll
    for (int nt = 0; nt < 2; ++nt) {
      v8f acc = (mt == 0) ? ((nt == 0) ? acc00 : acc01)
                          : ((nt == 0) ? acc10 : acc11);
      const int p = pin + wgx * 32 + nt * 16 + lr;
#pragma unroll
      for (int r = 0; r < 8; ++r) {
        const int mm = wgy * 32 + mt * 16 + hi * 8 + r;
        const float bwv = (mt == 0) ? bw0[r] : bw1[r];
        const size_t idx = ((size_t)bImg * Cn + mm) * HWn + p;
        out[idx] = acc[r] + bwv + x[idx];
      }
    }
  }
}

// ---------------------------------------------------------------------------
extern "C" void kernel_launch(void* const* d_in, const int* in_sizes, int n_in,
                              void* d_out, int out_size, void* d_ws, size_t ws_size,
                              hipStream_t stream) {
  const float* x    = (const float*)d_in[0];
  const float* wDw  = (const float*)d_in[1];
  const float* wOff = (const float*)d_in[2];
  const float* bOff = (const float*)d_in[3];
  const float* wPc  = (const float*)d_in[4];
  const float* wW   = (const float*)d_in[5];
  const float* bW   = (const float*)d_in[6];

  unsigned short* W2bf = (unsigned short*)d_ws;                       // 128 KiB
  float4* prm = (float4*)((char*)d_ws + (size_t)Cn * Cn * 2);         // 2 MiB

  cdcb_precompute_w2<<<Cn, Cn, 0, stream>>>(wW, wPc, wDw, W2bf);
  cdcb_offsets<<<(Bn * HWn) / 1024, 256, 0, stream>>>(x, wOff, bOff, prm);
  cdcb_main<<<(Bn * HWn) / 64, 512, 0, stream>>>(x, W2bf, prm, bW, (float*)d_out);
}
